// EfficientViTBlock_79559974191582
// MI455X (gfx1250) — matrix operand order repaired and checked by reading, not compiled, not run on hardware
//
#include <hip/hip_runtime.h>

// ---------------------------------------------------------------------------
// EfficientViT block for MI455X (gfx1250, wave32, WMMA bf16 16x16x32).
// All 1x1 convs / qkv / proj / attention matmuls run on v_wmma_f32_16x16x32_bf16
// with f32 accumulation. GEMM waves compute 32x32 register tiles (2x2 WMMA)
// for 2x fragment reuse. Depthwise 3x3 convs, BN, LN, softmax bookkeeping are
// f32 VALU. Canonical activation layout is token-major [T, C] (NHWC).
// ---------------------------------------------------------------------------

typedef __attribute__((ext_vector_type(16))) __bf16 v16bf;
typedef __attribute__((ext_vector_type(8)))  float  v8f;

#define T_TOK 16384   // 16 * 32 * 32 tokens
#define DIMC  256
#define MIDC  1024
#define HWSZ  1024
#define NHEAD 8
#define HD    32
#define EPSF  1e-5f

// ---------------- helpers ----------------

__device__ __forceinline__ unsigned short f2bf(float x) {
  unsigned int u = __float_as_uint(x);
  unsigned int r = (u + 0x7FFFu + ((u >> 16) & 1u)) >> 16;  // RNE
  return (unsigned short)r;
}
__device__ __forceinline__ float bf2f(unsigned short b) {
  return __uint_as_float(((unsigned int)b) << 16);
}

union FragBF { uint4 q[2]; v16bf v; };

// A-matrix fragment, 16x32 bf16, A row-major [M,K] (lda in elements).
// ISA layout: lane(0-15)=row, VGPR0-3 K=kh*8+0..7, VGPR4-7 K=16+kh*8+0..7,
// kh = lane>>4.  -> two contiguous 16B runs per lane.
__device__ __forceinline__ v16bf load_frag_a(const unsigned short* A, int lda,
                                             int m0, int kk, int lane) {
  int row = lane & 15, kh = lane >> 4;
  const unsigned short* p = A + (size_t)(m0 + row) * lda + kk + kh * 8;
  FragBF f;
  f.q[0] = *(const uint4*)(p);
  f.q[1] = *(const uint4*)(p + 16);
  return f.v;
}

// B-matrix fragment, 32x16 bf16, sourced from W [N,K] row-major so that
// B(:,col) = W[n0+col, kk..kk+31].  Lane: col = lane&15, K runs =
// kk + kh*8 + 0..7 and kk + 16 + kh*8 + 0..7 (kh = lane>>4), matching A.
__device__ __forceinline__ v16bf load_frag_b(const unsigned short* Wm, int ldw,
                                             int n0, int kk, int lane) {
  int col = lane & 15, koff = (lane >> 4) << 3;
  const unsigned short* p = Wm + (size_t)(n0 + col) * ldw + kk + koff;
  FragBF f;
  f.q[0] = *(const uint4*)(p);
  f.q[1] = *(const uint4*)(p + 16);
  return f.v;
}

__device__ __forceinline__ v8f wmma_bf16(v16bf a, v16bf b, v8f c) {
  return __builtin_amdgcn_wmma_f32_16x16x32_bf16(false, a, false, b,
                                                 (short)0, c, false, false);
}

__device__ __forceinline__ float wsum32(float v) {
  for (int off = 1; off < 32; off <<= 1) v += __shfl_xor(v, off, 32);
  return v;
}

// ---------------- generic WMMA GEMM with fused epilogue ----------------
// C[M,N] = act( bn( A[M,K] @ W[N,K]^T ) + bias ) + resid
// One wave computes one 32x32 block (2x2 WMMA tiles); 8 waves per block.
__global__ void k_gemm(const unsigned short* __restrict__ A, int lda,
                       const unsigned short* __restrict__ Wm, int ldw,
                       const float* __restrict__ bias,
                       const float* __restrict__ g, const float* __restrict__ be,
                       const float* __restrict__ mu, const float* __restrict__ va,
                       const float* __restrict__ resid, int ldr,
                       float* __restrict__ Cf, unsigned short* __restrict__ Cbf,
                       int ldc, int M, int N, int K, int do_silu) {
  int lane = threadIdx.x & 31;
  int wid  = blockIdx.x * (blockDim.x >> 5) + (threadIdx.x >> 5);
  int ntiles = N >> 5;                 // 32-wide tiles
  int total  = (M >> 5) * ntiles;
  if (wid >= total) return;
  int m0 = (wid / ntiles) << 5;
  int n0 = (wid % ntiles) << 5;

  // Warm the epilogue's residual stream (global_prefetch_b8).
  if (resid) __builtin_prefetch(resid + (size_t)(m0 + lane) * ldr + n0, 0, 1);

  v8f acc00 = {}, acc01 = {}, acc10 = {}, acc11 = {};
  for (int kk = 0; kk < K; kk += 32) {
    v16bf a0 = load_frag_a(A, lda, m0,      kk, lane);
    v16bf a1 = load_frag_a(A, lda, m0 + 16, kk, lane);
    v16bf b0 = load_frag_b(Wm, ldw, n0,      kk, lane);
    v16bf b1 = load_frag_b(Wm, ldw, n0 + 16, kk, lane);
    acc00 = wmma_bf16(a0, b0, acc00);
    acc01 = wmma_bf16(a0, b1, acc01);
    acc10 = wmma_bf16(a1, b0, acc10);
    acc11 = wmma_bf16(a1, b1, acc11);
  }

  v8f accs[2][2];
  accs[0][0] = acc00; accs[0][1] = acc01;
  accs[1][0] = acc10; accs[1][1] = acc11;
  int rb[2], cc[2];
  rb[0] = m0 + ((lane >> 4) << 3);
  rb[1] = rb[0] + 16;
  cc[0] = n0 + (lane & 15);
  cc[1] = cc[0] + 16;
  float invs[2] = {1.f, 1.f}, adds[2] = {0.f, 0.f};
  for (int j = 0; j < 2; ++j) {
    int col = cc[j];
    if (g) {
      float iv = g[col] * rsqrtf(va[col] + EPSF);
      invs[j] = iv;
      adds[j] = be[col] - mu[col] * iv;
    }
    if (bias) adds[j] += bias[col];
  }
  for (int i = 0; i < 2; ++i) {
    for (int j = 0; j < 2; ++j) {
      int col = cc[j];
      for (int r = 0; r < 8; ++r) {
        int row = rb[i] + r;
        float v = accs[i][j][r] * invs[j] + adds[j];
        if (do_silu) v = v / (1.f + __expf(-v));
        if (resid) v += resid[(size_t)row * ldr + col];
        if (Cf)  Cf [(size_t)row * ldc + col] = v;
        if (Cbf) Cbf[(size_t)row * ldc + col] = f2bf(v);
      }
    }
  }
}

// ---------------- layout transforms ----------------

__global__ void k_nchw_to_tok(const float* __restrict__ X, float* __restrict__ Y) {
  int gid = blockIdx.x * blockDim.x + threadIdx.x;     // T_TOK * DIMC
  int c = gid & (DIMC - 1);
  int t = gid >> 8;
  int b = t >> 10, hw = t & (HWSZ - 1);
  Y[(size_t)t * DIMC + c] = X[((size_t)b * DIMC + c) * HWSZ + hw];
}

__global__ void k_tok_to_nchw(const float* __restrict__ Y, float* __restrict__ X) {
  int gid = blockIdx.x * blockDim.x + threadIdx.x;
  int c = gid & (DIMC - 1);
  int t = gid >> 8;
  int b = t >> 10, hw = t & (HWSZ - 1);
  X[((size_t)b * DIMC + c) * HWSZ + hw] = Y[(size_t)t * DIMC + c];
}

__global__ void k_f32_to_bf16(const float* __restrict__ s,
                              unsigned short* __restrict__ d, int n) {
  int i = blockIdx.x * blockDim.x + threadIdx.x;
  if (i < n) d[i] = f2bf(s[i]);
}

// bn (per-channel, C=256) then cast to bf16, token-major in/out
__global__ void k_bn_bf16(const float* __restrict__ X,
                          const float* __restrict__ g, const float* __restrict__ be,
                          const float* __restrict__ mu, const float* __restrict__ va,
                          unsigned short* __restrict__ Y) {
  int gid = blockIdx.x * blockDim.x + threadIdx.x;     // T_TOK * DIMC
  int c = gid & (DIMC - 1);
  float iv = g[c] * rsqrtf(va[c] + EPSF);
  Y[gid] = f2bf(X[gid] * iv + (be[c] - mu[c] * iv));
}

// ---------------- depthwise 3x3 convs (NHWC) ----------------
// f32 input variant. Optional BN, bias, SiLU; writes f32 and/or bf16 with
// column offset into a wider [T, ldo] buffer (used for head concat).
__global__ void k_dw3x3_f32(const float* __restrict__ X,
                            const float* __restrict__ Wt,
                            const float* __restrict__ bias,
                            const float* __restrict__ g, const float* __restrict__ be,
                            const float* __restrict__ mu, const float* __restrict__ va,
                            float* __restrict__ Yf, unsigned short* __restrict__ Ybf,
                            int ldo, int ocol, int C, int do_silu) {
  long gid = (long)blockIdx.x * blockDim.x + threadIdx.x;
  if (gid >= (long)T_TOK * C) return;
  int  c = (int)(gid % C);
  long t = gid / C;
  int b = (int)(t >> 10), hw = (int)(t & (HWSZ - 1));
  int h = hw >> 5, ww = hw & 31;
  const float* wr = Wt + c * 9;
  float s = 0.f;
  for (int ky = 0; ky < 3; ++ky) {
    int hh = h + ky - 1;
    if (hh < 0 || hh >= 32) continue;
    for (int kx = 0; kx < 3; ++kx) {
      int w2 = ww + kx - 1;
      if (w2 < 0 || w2 >= 32) continue;
      s += X[((long)b * HWSZ + hh * 32 + w2) * C + c] * wr[ky * 3 + kx];
    }
  }
  if (g)    { float iv = g[c] * rsqrtf(va[c] + EPSF); s = s * iv + (be[c] - mu[c] * iv); }
  if (bias) s += bias[c];
  if (do_silu) s = s / (1.f + __expf(-s));
  long o = t * ldo + ocol + c;
  if (Yf)  Yf[o]  = s;
  if (Ybf) Ybf[o] = f2bf(s);
}

// bf16-input variant (FFN middle, C=1024): BN + SiLU, writes bf16.
__global__ void k_dw3x3_bf16(const unsigned short* __restrict__ X,
                             const float* __restrict__ Wt,
                             const float* __restrict__ g, const float* __restrict__ be,
                             const float* __restrict__ mu, const float* __restrict__ va,
                             unsigned short* __restrict__ Y, int C) {
  long gid = (long)blockIdx.x * blockDim.x + threadIdx.x;
  if (gid >= (long)T_TOK * C) return;
  int  c = (int)(gid % C);
  long t = gid / C;
  int b = (int)(t >> 10), hw = (int)(t & (HWSZ - 1));
  int h = hw >> 5, ww = hw & 31;
  const float* wr = Wt + c * 9;
  float s = 0.f;
  for (int ky = 0; ky < 3; ++ky) {
    int hh = h + ky - 1;
    if (hh < 0 || hh >= 32) continue;
    for (int kx = 0; kx < 3; ++kx) {
      int w2 = ww + kx - 1;
      if (w2 < 0 || w2 >= 32) continue;
      s += bf2f(X[((long)b * HWSZ + hh * 32 + w2) * C + c]) * wr[ky * 3 + kx];
    }
  }
  float iv = g[c] * rsqrtf(va[c] + EPSF);
  s = s * iv + (be[c] - mu[c] * iv);
  s = s / (1.f + __expf(-s));
  Y[t * C + c] = f2bf(s);
}

// ---------------- cascaded-head prep: bn + cascade-LN + in-LN -> bf16 ----
// One wave per token; lane = channel within head (HD == 32 == wave32).
__global__ void k_head_prep(const float* __restrict__ X1,
                            const float* __restrict__ OutAll,
                            const float* __restrict__ ang, const float* __restrict__ anb,
                            const float* __restrict__ anm, const float* __restrict__ anv,
                            const float* __restrict__ casw, const float* __restrict__ casb,
                            const float* __restrict__ inw,  const float* __restrict__ inb,
                            unsigned short* __restrict__ Hi, int head) {
  int lane = threadIdx.x & 31;
  int t = blockIdx.x * (blockDim.x >> 5) + (threadIdx.x >> 5);
  int c = head * HD + lane;
  float iv = ang[c] * rsqrtf(anv[c] + EPSF);
  float x = X1[(size_t)t * DIMC + c] * iv + (anb[c] - anm[c] * iv);
  if (casw) {  // cascade: x += LN(prev head output)
    float pv = OutAll[(size_t)t * DIMC + (head - 1) * HD + lane];
    float m  = wsum32(pv) * (1.f / 32.f);
    float d  = pv - m;
    float vv = wsum32(d * d) * (1.f / 32.f);
    x += d * rsqrtf(vv + EPSF) * casw[lane] + casb[lane];
  }
  float m2 = wsum32(x) * (1.f / 32.f);
  float d2 = x - m2;
  float v2 = wsum32(d2 * d2) * (1.f / 32.f);
  float h  = d2 * rsqrtf(v2 + EPSF) * inw[lane] + inb[lane];
  Hi[(size_t)t * HD + lane] = f2bf(h);
}

// V transpose for the attention PV matmul: Vt[b][d][n] = QKV[b*1024+n][64+d]
__global__ void k_vtrans(const unsigned short* __restrict__ QKV,
                         unsigned short* __restrict__ Vt) {
  int gid = blockIdx.x * blockDim.x + threadIdx.x;     // 16*32*1024
  int n = gid & (HWSZ - 1);
  int r = gid >> 10;
  int d = r & 31, b = r >> 5;
  Vt[((size_t)(b * HD + d)) * HWSZ + n] =
      QKV[((size_t)(b * HWSZ + n)) * 96 + 64 + d];
}

// ---------------- flash attention per head ----------------
// One wave per 16-query tile; online softmax over 1024 keys, 32 keys/step.
__global__ void k_attn(const unsigned short* __restrict__ QKV,   // [T,96] bf16
                       const unsigned short* __restrict__ Vt,    // [16][32][1024]
                       float* __restrict__ Ho) {                 // [T,32] f32
  __shared__ unsigned short plds[8][16][32];  // per-wave P transpose buffer
  int lane = threadIdx.x & 31;
  int widx = threadIdx.x >> 5;
  int tile = blockIdx.x * 8 + widx;           // 1024 tiles total
  int b  = tile >> 6;
  int q0 = (tile & 63) << 4;

  const unsigned short* Qb = QKV + (size_t)(b * HWSZ) * 96;
  const unsigned short* Kb = Qb + 32;
  const unsigned short* Vb = Vt + (size_t)b * HD * HWSZ;

  v16bf qf = load_frag_a(Qb, 96, q0, 0, lane);
  v8f o0 = {}, o1 = {};
  float mrow[8], lrow[8];
  for (int r = 0; r < 8; ++r) { mrow[r] = -1e30f; lrow[r] = 0.f; }
  const float scale = 0.17677669529663687f;   // 32^-0.5

  for (int kk = 0; kk < HWSZ; kk += 32) {
    v16bf kf0 = load_frag_b(Kb, 96, kk,      0, lane);
    v16bf kf1 = load_frag_b(Kb, 96, kk + 16, 0, lane);
    v8f s0 = {}, s1 = {};
    s0 = wmma_bf16(qf, kf0, s0);
    s1 = wmma_bf16(qf, kf1, s1);

    float p0[8], p1[8];
    for (int r = 0; r < 8; ++r) {
      float a = fminf(fmaxf(s0[r] * scale, -50.f), 50.f);
      float c = fminf(fmaxf(s1[r] * scale, -50.f), 50.f);
      float mx = fmaxf(a, c);
      for (int off = 1; off < 16; off <<= 1) mx = fmaxf(mx, __shfl_xor(mx, off, 32));
      float mn = fmaxf(mrow[r], mx);
      float alpha = __expf(mrow[r] - mn);
      float e0 = __expf(a - mn), e1 = __expf(c - mn);
      float rs = e0 + e1;
      for (int off = 1; off < 16; off <<= 1) rs += __shfl_xor(rs, off, 32);
      lrow[r] = lrow[r] * alpha + rs;
      mrow[r] = mn;
      o0[r] *= alpha;  o1[r] *= alpha;
      p0[r] = e0;      p1[r] = e1;
    }
    // C-layout -> A-layout for P via per-wave LDS tile
    __syncthreads();
    {
      int pr = (lane >> 4) << 3, pc = lane & 15;
      for (int r = 0; r < 8; ++r) {
        plds[widx][pr + r][pc]      = f2bf(p0[r]);
        plds[widx][pr + r][pc + 16] = f2bf(p1[r]);
      }
    }
    __syncthreads();
    v16bf pf  = load_frag_a(&plds[widx][0][0], 32, 0, 0, lane);
    v16bf vf0 = load_frag_b(Vb, HWSZ, 0,  kk, lane);   // dims 0-15
    v16bf vf1 = load_frag_b(Vb, HWSZ, 16, kk, lane);   // dims 16-31
    o0 = wmma_bf16(pf, vf0, o0);
    o1 = wmma_bf16(pf, vf1, o1);
  }

  int rbase = q0 + ((lane >> 4) << 3);
  int col = lane & 15;
  for (int r = 0; r < 8; ++r) {
    float invl = 1.f / lrow[r];
    size_t base = ((size_t)(b * HWSZ) + rbase + r) * HD;
    Ho[base + col]      = o0[r] * invl;
    Ho[base + 16 + col] = o1[r] * invl;
  }
}

// ---------------------------------------------------------------------------
// Host side
// ---------------------------------------------------------------------------

// d_in indices: setup_inputs() dict insertion order, depth-first.
enum {
  IN_X = 0,
  IN_LF_DW_W, IN_LF_DW_G, IN_LF_DW_B, IN_LF_DW_M, IN_LF_DW_V,
  IN_LF_PW_W, IN_LF_PW_G, IN_LF_PW_B, IN_LF_PW_M, IN_LF_PW_V,
  IN_AN_G, IN_AN_B, IN_AN_M, IN_AN_V,
  IN_ILN_W, IN_ILN_B,
  IN_QKV_W, IN_QKV_B,
  IN_HDW_W, IN_HDW_B,
  IN_CAS_W, IN_CAS_B,
  IN_PROJ_W, IN_PROJ_B,
  IN_FN_G, IN_FN_B, IN_FN_M, IN_FN_V,
  IN_PW1_W, IN_PW1_G, IN_PW1_B, IN_PW1_M, IN_PW1_V,
  IN_FDW_W, IN_FDW_G, IN_FDW_B, IN_FDW_M, IN_FDW_V,
  IN_PW2_W, IN_PW2_G, IN_PW2_B, IN_PW2_M, IN_PW2_V
};

static void gemm(hipStream_t s, const unsigned short* A, int lda,
                 const unsigned short* Wm, int ldw, const float* bias,
                 const float* g, const float* be, const float* mu, const float* va,
                 const float* resid, int ldr, float* Cf, unsigned short* Cbf,
                 int ldc, int M, int N, int K, int silu) {
  int total = (M / 32) * (N / 32);
  int blocks = (total + 7) / 8;
  k_gemm<<<blocks, 256, 0, s>>>(A, lda, Wm, ldw, bias, g, be, mu, va,
                                resid, ldr, Cf, Cbf, ldc, M, N, K, silu);
}

extern "C" void kernel_launch(void* const* d_in, const int* in_sizes, int n_in,
                              void* d_out, int out_size, void* d_ws, size_t ws_size,
                              hipStream_t stream) {
  auto F = [&](int i) { return (const float*)d_in[i]; };

  // -------- workspace layout --------
  char* w = (char*)d_ws;
  size_t off = 0;
  auto mk = [&](size_t bytes) -> char* {
    char* p = w + off;
    off += (bytes + 255) & ~(size_t)255;
    return p;
  };
  float* X0   = (float*)mk((size_t)T_TOK * DIMC * 4);   // input, token-major
  float* X1   = (float*)mk((size_t)T_TOK * DIMC * 4);   // after local ffn
  float* X2   = (float*)mk((size_t)T_TOK * DIMC * 4);   // after attention
  float* OutA = (float*)mk((size_t)T_TOK * DIMC * 4);   // head-concat (f32)
  float* OutT = (float*)mk((size_t)T_TOK * DIMC * 4);   // final, token-major
  unsigned short* A1    = (unsigned short*)mk((size_t)T_TOK * DIMC * 2);
  unsigned short* OutAb = (unsigned short*)mk((size_t)T_TOK * DIMC * 2);
  unsigned short* X2n   = (unsigned short*)mk((size_t)T_TOK * DIMC * 2);
  unsigned short* Y1    = (unsigned short*)mk((size_t)T_TOK * MIDC * 2);
  unsigned short* Y2    = (unsigned short*)mk((size_t)T_TOK * MIDC * 2);
  unsigned short* Hi    = (unsigned short*)mk((size_t)T_TOK * HD * 2);
  unsigned short* QKV   = (unsigned short*)mk((size_t)T_TOK * 96 * 2);
  unsigned short* Vt    = (unsigned short*)mk((size_t)16 * HD * HWSZ * 2);
  float*          Ho    = (float*)mk((size_t)T_TOK * HD * 4);
  unsigned short* Wpw   = (unsigned short*)mk((size_t)65536 * 2);
  unsigned short* Wqkv  = (unsigned short*)mk((size_t)NHEAD * 96 * HD * 2);
  unsigned short* Wproj = (unsigned short*)mk((size_t)65536 * 2);
  unsigned short* Wpw1  = (unsigned short*)mk((size_t)262144 * 2);
  unsigned short* Wpw2  = (unsigned short*)mk((size_t)262144 * 2);
  (void)ws_size; (void)in_sizes; (void)n_in;

  // -------- weight conversions (f32 -> bf16) --------
  k_f32_to_bf16<<<65536 / 256, 256, 0, stream>>>(F(IN_LF_PW_W), Wpw, 65536);
  k_f32_to_bf16<<<(NHEAD * 96 * HD + 255) / 256, 256, 0, stream>>>(F(IN_QKV_W), Wqkv, NHEAD * 96 * HD);
  k_f32_to_bf16<<<65536 / 256, 256, 0, stream>>>(F(IN_PROJ_W), Wproj, 65536);
  k_f32_to_bf16<<<262144 / 256, 256, 0, stream>>>(F(IN_PW1_W), Wpw1, 262144);
  k_f32_to_bf16<<<262144 / 256, 256, 0, stream>>>(F(IN_PW2_W), Wpw2, 262144);

  const int NE  = T_TOK * DIMC;      // 4,194,304 elements
  const int NEB = NE / 256;          // blocks of 256

  // -------- x: NCHW -> token-major --------
  k_nchw_to_tok<<<NEB, 256, 0, stream>>>(F(IN_X), X0);

  // -------- local ffn: dw3x3+bn+silu, pw(256->256)+bn+silu, +residual ------
  k_dw3x3_f32<<<NEB, 256, 0, stream>>>(X0, F(IN_LF_DW_W), nullptr,
      F(IN_LF_DW_G), F(IN_LF_DW_B), F(IN_LF_DW_M), F(IN_LF_DW_V),
      nullptr, A1, DIMC, 0, DIMC, 1);
  gemm(stream, A1, DIMC, Wpw, DIMC, nullptr,
       F(IN_LF_PW_G), F(IN_LF_PW_B), F(IN_LF_PW_M), F(IN_LF_PW_V),
       X0, DIMC, X1, nullptr, DIMC, T_TOK, DIMC, DIMC, 1);

  // -------- cascaded group attention (heads sequential) --------
  for (int h = 0; h < NHEAD; ++h) {
    const float* casw = (h > 0) ? F(IN_CAS_W) + (h - 1) * HD : nullptr;
    const float* casb = (h > 0) ? F(IN_CAS_B) + (h - 1) * HD : nullptr;
    k_head_prep<<<T_TOK / 8, 256, 0, stream>>>(X1, OutA,
        F(IN_AN_G), F(IN_AN_B), F(IN_AN_M), F(IN_AN_V),
        casw, casb, F(IN_ILN_W), F(IN_ILN_B), Hi, h);
    // qkv: [T,32] @ [96,32]^T + b -> bf16 [T,96]
    gemm(stream, Hi, HD, Wqkv + (size_t)h * 96 * HD, HD, F(IN_QKV_B) + h * 96,
         nullptr, nullptr, nullptr, nullptr, nullptr, 0,
         nullptr, QKV, 96, T_TOK, 96, HD, 0);
    k_vtrans<<<(16 * HD * HWSZ) / 256, 256, 0, stream>>>(QKV, Vt);
    k_attn<<<128, 256, 0, stream>>>(QKV, Vt, Ho);
    // per-head dw3x3 + bias, write into concat buffer columns [h*32, h*32+32)
    k_dw3x3_f32<<<(T_TOK * HD) / 256, 256, 0, stream>>>(Ho,
        F(IN_HDW_W) + h * HD * 9, F(IN_HDW_B) + h * HD,
        nullptr, nullptr, nullptr, nullptr,
        OutA, OutAb, DIMC, h * HD, HD, 0);
  }
  // proj (256->256) + bias, residual X1 -> X2
  gemm(stream, OutAb, DIMC, Wproj, DIMC, F(IN_PROJ_B),
       nullptr, nullptr, nullptr, nullptr,
       X1, DIMC, X2, nullptr, DIMC, T_TOK, DIMC, DIMC, 0);

  // -------- ffn: bn, pw1(256->1024)+bn+silu, dw3x3+bn+silu, pw2+bn+silu ----
  k_bn_bf16<<<NEB, 256, 0, stream>>>(X2, F(IN_FN_G), F(IN_FN_B), F(IN_FN_M),
                                     F(IN_FN_V), X2n);
  gemm(stream, X2n, DIMC, Wpw1, DIMC, nullptr,
       F(IN_PW1_G), F(IN_PW1_B), F(IN_PW1_M), F(IN_PW1_V),
       nullptr, 0, nullptr, Y1, MIDC, T_TOK, MIDC, DIMC, 1);
  k_dw3x3_bf16<<<(T_TOK * MIDC) / 256, 256, 0, stream>>>(Y1, F(IN_FDW_W),
      F(IN_FDW_G), F(IN_FDW_B), F(IN_FDW_M), F(IN_FDW_V), Y2, MIDC);
  gemm(stream, Y2, MIDC, Wpw2, MIDC, nullptr,
       F(IN_PW2_G), F(IN_PW2_B), F(IN_PW2_M), F(IN_PW2_V),
       X2, DIMC, OutT, nullptr, DIMC, T_TOK, DIMC, MIDC, 1);

  // -------- back to NCHW --------
  k_tok_to_nchw<<<NEB, 256, 0, stream>>>(OutT, (float*)d_out);
}
